// TransformerSeqLayer_84370337563147
// MI455X (gfx1250) — matrix operand order repaired and checked by reading, not compiled, not run on hardware
//
#include <hip/hip_runtime.h>
#include <cstdint>
#include <cstddef>

// ---------------------------------------------------------------------------
// TransformerSeqLayer for MI455X (gfx1250): bf16 WMMA GEMMs + async-to-LDS
// B=4, M=1024, SPAN=1024, D=512, H=8, HD=64, DFF=2048, E=8, TOPK=2
// ---------------------------------------------------------------------------

using u16 = unsigned short;                 // bf16 bit container
#define DEV __device__ __forceinline__

typedef __bf16 v16bf __attribute__((ext_vector_type(16)));
typedef float  v8f   __attribute__((ext_vector_type(8)));

struct Frag { int4 x, y; };                 // 32B = 16 bf16 per lane

DEV v16bf as_bf16(Frag f) { return __builtin_bit_cast(v16bf, f); }

DEV u16 f2bf(float f) {
  unsigned u = __builtin_bit_cast(unsigned, f);
  return (u16)((u + 0x7fffu + ((u >> 16) & 1u)) >> 16);   // RNE
}
DEV float bf2f(u16 h) {
  unsigned u = ((unsigned)h) << 16;
  return __builtin_bit_cast(float, u);
}

// ---- CDNA5 async global->LDS copy (16B per lane), tracked by ASYNCcnt ----
DEV void async_b128(void* lds, const void* g) {
  unsigned off = (unsigned)(unsigned long long)lds;  // flat LDS addr[31:0] == LDS offset
  asm volatile("global_load_async_to_lds_b128 %0, %1, off"
               :: "v"(off), "v"(g) : "memory");
}
DEV void s_wait_async8() { asm volatile("s_wait_asynccnt 8" ::: "memory"); }
DEV void s_wait_async0() { asm volatile("s_wait_asynccnt 0" ::: "memory"); }

// ---- WMMA fragment loads (ISA 7.12.2 layouts) ----
// A 16x32 bf16: lanes0-15 row=l, K pairs v0..3:K0-7, v4..7:K16-23; hi-lanes +8
DEV Frag load_fragA(const u16* base, int stride, int row, int k0, int hi) {
  const char* p = (const char*)(base + (size_t)row * stride + k0);
  Frag f;
  f.x = *(const int4*)(p + hi * 16);
  f.y = *(const int4*)(p + 32 + hi * 16);
  return f;
}
// B 32x16 bf16 from [col][K]-major LDS: lanes0-15 K=0-7,16-23; lanes16-31 K=8-15,24-31
DEV Frag load_fragB(const u16* base, int stride, int col, int k0, int hi) {
  const char* p = (const char*)(base + (size_t)col * stride + k0) + hi * 16;
  Frag f;
  f.x = *(const int4*)p;
  f.y = *(const int4*)(p + 32);
  return f;
}
DEV v8f wmma_bf16(Frag a, Frag b, v8f c) {
  return __builtin_amdgcn_wmma_f32_16x16x32_bf16(false, as_bf16(a),
                                                 false, as_bf16(b),
                                                 (short)0, c, false, false);
}

DEV float wsum32(float v) {
#pragma unroll
  for (int m = 16; m; m >>= 1) v += __shfl_xor(v, m, 32);
  return v;
}
DEV float wmax16(float v) {
#pragma unroll
  for (int m = 8; m; m >>= 1) v = fmaxf(v, __shfl_xor(v, m, 32));
  return v;
}
DEV float wsum16(float v) {
#pragma unroll
  for (int m = 8; m; m >>= 1) v += __shfl_xor(v, m, 32);
  return v;
}

// ===========================================================================
// Weight transpose + f32->bf16:  in [R][C] f32  ->  out [C][R] bf16
// ===========================================================================
__global__ __launch_bounds__(256) void tr_cast_k(const float* __restrict__ in,
                                                 u16* __restrict__ out,
                                                 int R, int C, long iz, long oz) {
  __shared__ float t[32][33];
  int z = blockIdx.z;
  const float* I = in + (long)z * iz;
  u16* O = out + (long)z * oz;
  int c0 = blockIdx.x * 32, r0 = blockIdx.y * 32;
#pragma unroll
  for (int i = 0; i < 4; i++) {
    int r = r0 + threadIdx.y + i * 8, c = c0 + threadIdx.x;
    if (r < R && c < C) t[threadIdx.y + i * 8][threadIdx.x] = I[(long)r * C + c];
  }
  __syncthreads();
#pragma unroll
  for (int i = 0; i < 4; i++) {
    int c = c0 + threadIdx.y + i * 8, r = r0 + threadIdx.x;
    if (r < R && c < C) O[(long)c * R + r] = f2bf(t[threadIdx.x][threadIdx.y + i * 8]);
  }
}

// hall[b][0..2048)[512] bf16 = concat(h_cache, h) per batch
__global__ __launch_bounds__(256) void concat_cast_k(const float* __restrict__ h,
                                                     const float* __restrict__ hc,
                                                     u16* __restrict__ hall) {
  long i = (long)blockIdx.x * 256 + threadIdx.x;   // < 4*2048*512
  int c = (int)(i & 511);
  long r = i >> 9;
  int j = (int)(r & 2047), b = (int)(r >> 11);
  float v = (j < 1024) ? hc[((long)b * 1024 + j) * 512 + c]
                       : h[((long)b * 1024 + (j - 1024)) * 512 + c];
  hall[i] = f2bf(v);
}

// ===========================================================================
// Generic bf16 WMMA GEMM: C[M,N] = A[M,K] @ W[K,N], W pre-transposed [N][K].
// 128 threads (4 waves), 64x64 tile, K-block 64, async double-buffered LDS.
// ===========================================================================
enum : int { EPI_BF16 = 0, EPI_HEADS, EPI_HEADS_T, EPI_F32, EPI_RELU, EPI_MOE2 };

struct GemmP {
  const u16* A; const u16* WT; const float* bias;
  u16* outB; float* outF; float* yacc;
  const int* a_rowmap; const int* a_dynoff; const int* cnt;
  const int* o_dynoff; const int* rowmap; const float* wslot;
  long aZ, wZ, oZ;
  int M, N, K, bZ, Rrows;
};

template <int EPI>
__global__ __launch_bounds__(128) void gemm_k(GemmP P) {
  __shared__ u16 As[2][64][72];
  __shared__ u16 Ws[2][64][72];
  const int tid = threadIdx.x, wv = tid >> 5, lane = tid & 31;
  const int hi = lane >> 4, ln = lane & 15;
  const int z = blockIdx.z;
  const int m0 = blockIdx.y * 64, n0 = blockIdx.x * 64;
  if (P.cnt && m0 >= P.cnt[z]) return;     // uniform early-exit (MoE)

  auto stage = [&](int buf, int k0) {
#pragma unroll
    for (int i = 0; i < 4; i++) {
      int c = i * 128 + tid;
      int r = c >> 3, q8 = (c & 7) * 8;
      long srow;
      {
        int mr = m0 + r;
        if (P.cnt) { int cm = P.cnt[z]; if (mr > cm - 1) mr = cm - 1; }
        if (P.a_rowmap)      srow = P.a_rowmap[z * 4096 + mr];
        else if (P.a_dynoff) srow = (long)P.a_dynoff[z] + mr;
        else                 srow = mr;
      }
      async_b128(&As[buf][r][q8], P.A + (long)z * P.aZ + srow * (long)P.K + k0 + q8);
      async_b128(&Ws[buf][r][q8], P.WT + (long)z * P.wZ + (long)(n0 + r) * P.K + k0 + q8);
    }
  };

  v8f acc[4] = {};
  const int nK = P.K >> 6;
  stage(0, 0);
  for (int kb = 0; kb < nK; ++kb) {
    if (kb + 1 < nK) { stage((kb + 1) & 1, (kb + 1) << 6); s_wait_async8(); }
    else             { s_wait_async0(); }
    __syncthreads();
    const u16* Ab = &As[kb & 1][wv * 16][0];
    const u16* Wb = &Ws[kb & 1][0][0];
#pragma unroll
    for (int kc = 0; kc < 2; kc++) {
      Frag a = load_fragA(Ab, 72, ln, kc * 32, hi);
#pragma unroll
      for (int nt = 0; nt < 4; nt++) {
        Frag b = load_fragB(Wb, 72, nt * 16 + ln, kc * 32, hi);
        acc[nt] = wmma_bf16(a, b, acc[nt]);
      }
    }
    __syncthreads();
  }

  // epilogue: element (m, n); C layout: vgpr r -> row r (+8 hi half), col = lane&15
#pragma unroll
  for (int nt = 0; nt < 4; nt++) {
#pragma unroll
    for (int r = 0; r < 8; r++) {
      int m = m0 + wv * 16 + r + hi * 8;
      int n = n0 + nt * 16 + ln;
      float v = acc[nt][r];
      if constexpr (EPI == EPI_F32) {
        float bb = P.bias ? P.bias[n] : 0.f;
        P.outF[(long)m * P.N + n] = v + bb;
      } else if constexpr (EPI == EPI_BF16) {
        P.outB[(long)z * P.oZ + (long)m * P.N + n] = f2bf(v);
      } else if constexpr (EPI == EPI_HEADS) {
        int hh = n >> 6, d = n & 63;
        P.outB[((long)(z * 8 + hh) * P.Rrows + m) * 64 + d] = f2bf(v);
      } else if constexpr (EPI == EPI_HEADS_T) {
        int hh = n >> 6, d = n & 63;
        P.outB[((long)(z * 8 + hh) * 64 + d) * P.Rrows + m] = f2bf(v);
      } else if constexpr (EPI == EPI_RELU) {
        if (!P.cnt || m < P.cnt[z]) {
          float rv = v + P.bias[(long)z * P.bZ + n];
          rv = rv > 0.f ? rv : 0.f;
          long dr = (P.o_dynoff ? P.o_dynoff[z] : 0) + m;
          P.outB[dr * P.N + n] = f2bf(rv);
        }
      } else if constexpr (EPI == EPI_MOE2) {
        if (m < P.cnt[z]) {
          int tok = P.rowmap[z * 4096 + m];
          float w = P.wslot[z * 4096 + m];
          atomicAdd(&P.yacc[(long)tok * 512 + n], w * (v + P.bias[(long)z * P.bZ + n]));
        }
      }
    }
  }
}

// ===========================================================================
// Windowed rel-pos attention, flash-style. Per WG: one bh, 64 query rows
// (4 waves x 16). Keys n in [m0, m0+1088) in 17 blocks of 64; valid iff
// 0 <= n-m < SPAN. K/V tiles async double-buffered; POS gathered from global.
// ===========================================================================
__global__ __launch_bounds__(128) void attn_k(const u16* __restrict__ qh,
                                              const u16* __restrict__ kh,
                                              const u16* __restrict__ vT,
                                              const u16* __restrict__ POS,
                                              u16* __restrict__ attn_o) {
  __shared__ u16 Kt[2][64][72];   // [key][d]
  __shared__ u16 Vt[2][64][72];   // [d][key]
  __shared__ u16 Ps[4][16][72];   // per-wave P staging
  const int tid = threadIdx.x, wv = tid >> 5, lane = tid & 31;
  const int hi = lane >> 4, ln = lane & 15;
  const int bh = blockIdx.y, b = bh >> 3, h = bh & 7;
  const int m0 = blockIdx.x * 64, mw0 = m0 + wv * 16;
  const float scale = 0.044194173824159216f;  // 1/sqrt(512)

  Frag qA[2];
  {
    const u16* qb = qh + ((long)bh * 1024 + mw0) * 64;
#pragma unroll
    for (int kc = 0; kc < 2; kc++) qA[kc] = load_fragA(qb, 64, ln, kc * 32, hi);
  }
  v8f acc[4] = {};
  float mst[8], lst[8];
#pragma unroll
  for (int r = 0; r < 8; r++) { mst[r] = -1e30f; lst[r] = 0.f; }

  auto stageKV = [&](int buf, int nb) {
#pragma unroll
    for (int i = 0; i < 4; i++) {
      int c = i * 128 + tid;
      int r = c >> 3, q8 = (c & 7) * 8;
      async_b128(&Kt[buf][r][q8], kh + ((long)bh * 2048 + nb + r) * 64 + q8);
      async_b128(&Vt[buf][r][q8], vT + ((long)bh * 64 + r) * 2048 + nb + q8);
    }
  };

  stageKV(0, m0);
  for (int j = 0; j < 17; ++j) {
    const int n0 = m0 + (j << 6);
    if (j + 1 < 17) { stageKV((j + 1) & 1, n0 + 64); s_wait_async8(); }
    else            { s_wait_async0(); }
    __syncthreads();
    const int buf = j & 1;

    // scores = (q.k + POS[m][n-m]) * scale, masked to window
    float S[4][8];
#pragma unroll
    for (int nt = 0; nt < 4; nt++) {
      v8f sc = {};
#pragma unroll
      for (int kc = 0; kc < 2; kc++) {
        Frag kb = load_fragB(&Kt[buf][0][0], 72, nt * 16 + ln, kc * 32, hi);
        sc = wmma_bf16(qA[kc], kb, sc);
      }
#pragma unroll
      for (int r = 0; r < 8; r++) {
        int m = mw0 + r + hi * 8;
        int n = n0 + nt * 16 + ln;
        int l = n - m;
        float s = -1e30f;
        if (l >= 0 && l < 1024)
          s = (sc[r] + bf2f(POS[((long)bh * 1024 + m) * 1024 + l])) * scale;
        S[nt][r] = s;
      }
    }
    // online softmax update (stats replicated across the 16 lanes of a row)
    float al[8];
#pragma unroll
    for (int r = 0; r < 8; r++) {
      float bm = S[0][r];
#pragma unroll
      for (int nt = 1; nt < 4; nt++) bm = fmaxf(bm, S[nt][r]);
      bm = wmax16(bm);
      float mn = fmaxf(mst[r], bm);
      al[r] = __expf(mst[r] - mn);
      mst[r] = mn;
    }
#pragma unroll
    for (int r = 0; r < 8; r++) {
      float rs = 0.f;
#pragma unroll
      for (int nt = 0; nt < 4; nt++) {
        float p = __expf(S[nt][r] - mst[r]);
        S[nt][r] = p;
        rs += p;
      }
      lst[r] = lst[r] * al[r] + wsum16(rs);
    }
#pragma unroll
    for (int dt = 0; dt < 4; dt++)
#pragma unroll
      for (int r = 0; r < 8; r++) acc[dt][r] *= al[r];

    // P -> LDS (bf16) -> A-fragments, then acc += P @ V
#pragma unroll
    for (int nt = 0; nt < 4; nt++)
#pragma unroll
      for (int r = 0; r < 8; r++)
        Ps[wv][r + hi * 8][nt * 16 + ln] = f2bf(S[nt][r]);
    Frag pA[2];
#pragma unroll
    for (int kc = 0; kc < 2; kc++) pA[kc] = load_fragA(&Ps[wv][0][0], 72, ln, kc * 32, hi);
#pragma unroll
    for (int dt = 0; dt < 4; dt++)
#pragma unroll
      for (int kc = 0; kc < 2; kc++) {
        Frag vb = load_fragB(&Vt[buf][0][0], 72, dt * 16 + ln, kc * 32, hi);
        acc[dt] = wmma_bf16(pA[kc], vb, acc[dt]);
      }
    __syncthreads();
  }

  // out[b][m][h*64+d] bf16 (merged-head layout for the Wo GEMM)
#pragma unroll
  for (int dt = 0; dt < 4; dt++)
#pragma unroll
    for (int r = 0; r < 8; r++) {
      int m = mw0 + r + hi * 8;
      int d = dt * 16 + ln;
      attn_o[((long)(b * 1024 + m)) * 512 + h * 64 + d] = f2bf(acc[dt][r] / lst[r]);
    }
}

// ===========================================================================
// LayerNorm kernels (one wave per row, 4 rows per 128-thread block)
// ===========================================================================
DEV void ln_store(const float x[16], const float* g, const float* bb, int lane,
                  long off, float* of, u16* ob) {
  float mu = 0.f;
#pragma unroll
  for (int i = 0; i < 16; i++) mu += x[i];
  mu = wsum32(mu) * (1.f / 512.f);
  float var = 0.f;
#pragma unroll
  for (int i = 0; i < 16; i++) { float d = x[i] - mu; var += d * d; }
  var = wsum32(var) * (1.f / 512.f);
  float rs = rsqrtf(var + 1e-5f);
#pragma unroll
  for (int i = 0; i < 16; i++) {
    int c = lane + 32 * i;
    float y = (x[i] - mu) * rs * g[c] + bb[c];
    if (of) of[off + c] = y;
    if (ob) ob[off + c] = f2bf(y);
  }
}

__global__ __launch_bounds__(128) void ln1_k(const float* h, const float* wo,
                                             const float* g, const float* bb,
                                             float* h1f, u16* h1b) {
  int row = blockIdx.x * 4 + (threadIdx.x >> 5), lane = threadIdx.x & 31;
  long off = (long)row * 512;
  float x[16];
#pragma unroll
  for (int i = 0; i < 16; i++) { int c = lane + 32 * i; x[i] = h[off + c] + wo[off + c]; }
  ln_store(x, g, bb, lane, off, h1f, h1b);
}

__global__ __launch_bounds__(128) void ln2_k(const float* h1f, const float* y,
                                             const float* mom, const float* g,
                                             const float* bb, float* outMom,
                                             float* h2f, u16* h2b) {
  int row = blockIdx.x * 4 + (threadIdx.x >> 5), lane = threadIdx.x & 31;
  long off = (long)row * 512;
  float x[16];
#pragma unroll
  for (int i = 0; i < 16; i++) {
    int c = lane + 32 * i;
    float mn = 0.7f * mom[off + c] + y[off + c];   // MU=0.7, GAMMA=1.0
    outMom[off + c] = mn;
    x[i] = 2.f * h1f[off + c] - mn;                // h1 + (h1 - mn)
  }
  ln_store(x, g, bb, lane, off, h2f, h2b);
}

__global__ __launch_bounds__(128) void ln3_k(const float* h2f, const float* ff,
                                             const float* g, const float* bb,
                                             float* out) {
  int row = blockIdx.x * 4 + (threadIdx.x >> 5), lane = threadIdx.x & 31;
  long off = (long)row * 512;
  float x[16];
#pragma unroll
  for (int i = 0; i < 16; i++) { int c = lane + 32 * i; x[i] = h2f[off + c] + ff[off + c]; }
  ln_store(x, g, bb, lane, off, out, nullptr);
}

// ===========================================================================
// MoE gate: logits, top-2 softmax, per-expert row lists
// ===========================================================================
__global__ __launch_bounds__(32) void gate_k(const float* __restrict__ h1f,
                                             const float* __restrict__ Wg,
                                             const float* __restrict__ bg,
                                             int* cnt, int* rowmap, float* wslot) {
  int tok = blockIdx.x, lane = threadIdx.x;
  float s[8] = {};
#pragma unroll
  for (int i = 0; i < 16; i++) {
    int c = lane + 32 * i;
    float x = h1f[(long)tok * 512 + c];
    const float* wr = Wg + (long)c * 8;
#pragma unroll
    for (int e = 0; e < 8; e++) s[e] += x * wr[e];
  }
#pragma unroll
  for (int e = 0; e < 8; e++) s[e] = wsum32(s[e]);
  if (lane == 0) {
#pragma unroll
    for (int e = 0; e < 8; e++) s[e] += bg[e];
    int i1 = 0;
#pragma unroll
    for (int e = 1; e < 8; e++) if (s[e] > s[i1]) i1 = e;
    int i2 = (i1 == 0) ? 1 : 0;
#pragma unroll
    for (int e = 0; e < 8; e++) if (e != i1 && s[e] > s[i2]) i2 = e;
    float e2 = __expf(s[i2] - s[i1]);
    float p1 = 1.f / (1.f + e2), p2 = e2 * p1;
    int a = atomicAdd(&cnt[i1], 1);
    rowmap[i1 * 4096 + a] = tok; wslot[i1 * 4096 + a] = p1;
    int c2 = atomicAdd(&cnt[i2], 1);
    rowmap[i2 * 4096 + c2] = tok; wslot[i2 * 4096 + c2] = p2;
  }
}

__global__ void offs_k(const int* cnt, int* offs) {
  if (threadIdx.x == 0 && blockIdx.x == 0) {
    int run = 0;
    for (int e = 0; e < 8; e++) { offs[e] = run; run += cnt[e]; }
  }
}

// ===========================================================================
// Host launcher
// ===========================================================================
extern "C" void kernel_launch(void* const* d_in, const int* in_sizes, int n_in,
                              void* d_out, int out_size, void* d_ws, size_t ws_size,
                              hipStream_t stream) {
  const float* h   = (const float*)d_in[0];
  const float* hc  = (const float*)d_in[1];
  const float* pos = (const float*)d_in[2];
  const float* mom = (const float*)d_in[3];
  const float* Wq  = (const float*)d_in[4];
  const float* Wk  = (const float*)d_in[5];
  const float* Wv  = (const float*)d_in[6];
  const float* Wo  = (const float*)d_in[7];
  const float* Wg  = (const float*)d_in[8];
  const float* bg  = (const float*)d_in[9];
  const float* W1  = (const float*)d_in[10];
  const float* b1  = (const float*)d_in[11];
  const float* W2  = (const float*)d_in[12];
  const float* b2  = (const float*)d_in[13];
  const float* Wf1 = (const float*)d_in[14];
  const float* bf1 = (const float*)d_in[15];
  const float* Wf2 = (const float*)d_in[16];
  const float* bf2 = (const float*)d_in[17];
  const float* l1g = (const float*)d_in[18];
  const float* l1b = (const float*)d_in[19];
  const float* l2g = (const float*)d_in[20];
  const float* l2b = (const float*)d_in[21];
  const float* l3g = (const float*)d_in[22];
  const float* l3b = (const float*)d_in[23];
  float* out_h3  = (float*)d_out;
  float* out_mom = out_h3 + (size_t)4096 * 512;

  char* wsb = (char*)d_ws; size_t cur = 0;
  auto alloc = [&](size_t bytes) -> void* {
    cur = (cur + 255) & ~(size_t)255;
    void* p = wsb + cur; cur += bytes; return p;
  };
  u16* hall  = (u16*)alloc((size_t)4 * 2048 * 512 * 2);
  u16* WqT   = (u16*)alloc((size_t)512 * 512 * 2);
  u16* WkT   = (u16*)alloc((size_t)512 * 512 * 2);
  u16* WvT   = (u16*)alloc((size_t)512 * 512 * 2);
  u16* WoT   = (u16*)alloc((size_t)512 * 512 * 2);
  u16* W1T   = (u16*)alloc((size_t)8 * 2048 * 512 * 2);
  u16* W2T   = (u16*)alloc((size_t)8 * 512 * 2048 * 2);
  u16* Wf1T  = (u16*)alloc((size_t)2048 * 512 * 2);
  u16* Wf2T  = (u16*)alloc((size_t)512 * 2048 * 2);
  u16* posT  = (u16*)alloc((size_t)1024 * 64 * 2);
  u16* qhB   = (u16*)alloc((size_t)32 * 1024 * 64 * 2);
  u16* khB   = (u16*)alloc((size_t)32 * 2048 * 64 * 2);
  u16* vTB   = (u16*)alloc((size_t)32 * 2048 * 64 * 2);
  u16* POSb  = (u16*)alloc((size_t)32 * 1024 * 1024 * 2);
  u16* attnO = (u16*)alloc((size_t)4096 * 512 * 2);
  float* woO = (float*)alloc((size_t)4096 * 512 * 4);
  float* h1f = (float*)alloc((size_t)4096 * 512 * 4);
  u16* h1b   = (u16*)alloc((size_t)4096 * 512 * 2);
  int* cnt    = (int*)alloc(256);
  int* offs   = (int*)alloc(256);
  int* rowmap = (int*)alloc((size_t)8 * 4096 * 4);
  float* wsl  = (float*)alloc((size_t)8 * 4096 * 4);
  u16* act    = (u16*)alloc((size_t)8192 * 2048 * 2);
  float* yb   = (float*)alloc((size_t)4096 * 512 * 4);
  float* h2f  = (float*)alloc((size_t)4096 * 512 * 4);
  u16* h2b    = (u16*)alloc((size_t)4096 * 512 * 2);
  u16* actf   = (u16*)alloc((size_t)4096 * 2048 * 2);
  float* ffb  = (float*)alloc((size_t)4096 * 512 * 4);

  dim3 tb(32, 8);
  // ---- weight prep: transpose + cast to bf16 ----
  tr_cast_k<<<dim3(16, 16, 1), tb, 0, stream>>>(Wq, WqT, 512, 512, 0, 0);
  tr_cast_k<<<dim3(16, 16, 1), tb, 0, stream>>>(Wk, WkT, 512, 512, 0, 0);
  tr_cast_k<<<dim3(16, 16, 1), tb, 0, stream>>>(Wv, WvT, 512, 512, 0, 0);
  tr_cast_k<<<dim3(16, 16, 1), tb, 0, stream>>>(Wo, WoT, 512, 512, 0, 0);
  tr_cast_k<<<dim3(64, 16, 8), tb, 0, stream>>>(W1, W1T, 512, 2048, 512L * 2048, 512L * 2048);
  tr_cast_k<<<dim3(16, 64, 8), tb, 0, stream>>>(W2, W2T, 2048, 512, 2048L * 512, 2048L * 512);
  tr_cast_k<<<dim3(64, 16, 1), tb, 0, stream>>>(Wf1, Wf1T, 512, 2048, 0, 0);
  tr_cast_k<<<dim3(16, 64, 1), tb, 0, stream>>>(Wf2, Wf2T, 2048, 512, 0, 0);
  tr_cast_k<<<dim3(32, 2, 1), tb, 0, stream>>>(pos, posT, 64, 1024, 0, 0);
  concat_cast_k<<<16384, 256, 0, stream>>>(h, hc, hall);

  // ---- q/k/v projections ----
  { GemmP P{}; P.A = hall + 1024L * 512; P.WT = WqT; P.outB = qhB;
    P.M = 1024; P.N = 512; P.K = 512; P.aZ = 2048L * 512; P.Rrows = 1024;
    gemm_k<EPI_HEADS><<<dim3(8, 16, 4), 128, 0, stream>>>(P); }
  { GemmP P{}; P.A = hall; P.WT = WkT; P.outB = khB;
    P.M = 2048; P.N = 512; P.K = 512; P.aZ = 2048L * 512; P.Rrows = 2048;
    gemm_k<EPI_HEADS><<<dim3(8, 32, 4), 128, 0, stream>>>(P); }
  { GemmP P{}; P.A = hall; P.WT = WvT; P.outB = vTB;
    P.M = 2048; P.N = 512; P.K = 512; P.aZ = 2048L * 512; P.Rrows = 2048;
    gemm_k<EPI_HEADS_T><<<dim3(8, 32, 4), 128, 0, stream>>>(P); }
  // ---- positional term: POS[bh][m][l] = q . pos[:,l] ----
  { GemmP P{}; P.A = qhB; P.WT = posT; P.outB = POSb;
    P.M = 1024; P.N = 1024; P.K = 64; P.aZ = 1024L * 64; P.oZ = 1024L * 1024;
    gemm_k<EPI_BF16><<<dim3(16, 16, 32), 128, 0, stream>>>(P); }
  // ---- attention ----
  attn_k<<<dim3(16, 32), 128, 0, stream>>>(qhB, khB, vTB, POSb, attnO);
  // ---- Wo + residual LN1 ----
  { GemmP P{}; P.A = attnO; P.WT = WoT; P.outF = woO;
    P.M = 4096; P.N = 512; P.K = 512;
    gemm_k<EPI_F32><<<dim3(8, 64, 1), 128, 0, stream>>>(P); }
  ln1_k<<<1024, 128, 0, stream>>>(h, woO, l1g, l1b, h1f, h1b);
  // ---- MoE ----
  hipMemsetAsync(cnt, 0, 8 * sizeof(int), stream);
  hipMemsetAsync(yb, 0, (size_t)4096 * 512 * 4, stream);
  gate_k<<<4096, 32, 0, stream>>>(h1f, Wg, bg, cnt, rowmap, wsl);
  offs_k<<<1, 1, 0, stream>>>(cnt, offs);
  { GemmP P{}; P.A = h1b; P.WT = W1T; P.bias = b1; P.outB = act;
    P.M = 4096; P.N = 2048; P.K = 512; P.wZ = 2048L * 512; P.bZ = 2048;
    P.a_rowmap = rowmap; P.cnt = cnt; P.o_dynoff = offs;
    gemm_k<EPI_RELU><<<dim3(32, 64, 8), 128, 0, stream>>>(P); }
  { GemmP P{}; P.A = act; P.WT = W2T; P.bias = b2; P.yacc = yb;
    P.M = 4096; P.N = 512; P.K = 2048; P.wZ = 512L * 2048; P.bZ = 512;
    P.a_dynoff = offs; P.cnt = cnt; P.rowmap = rowmap; P.wslot = wsl;
    gemm_k<EPI_MOE2><<<dim3(8, 64, 8), 128, 0, stream>>>(P); }
  ln2_k<<<1024, 128, 0, stream>>>(h1f, yb, mom, l2g, l2b, out_mom, h2f, h2b);
  // ---- dense FFN + LN3 ----
  { GemmP P{}; P.A = h2b; P.WT = Wf1T; P.bias = bf1; P.outB = actf;
    P.M = 4096; P.N = 2048; P.K = 512;
    gemm_k<EPI_RELU><<<dim3(32, 64, 1), 128, 0, stream>>>(P); }
  { GemmP P{}; P.A = actf; P.WT = Wf2T; P.bias = bf2; P.outF = ffb;
    P.M = 4096; P.N = 512; P.K = 2048;
    gemm_k<EPI_F32><<<dim3(8, 64, 1), 128, 0, stream>>>(P); }
  ln3_k<<<1024, 128, 0, stream>>>(h2f, ffb, l3g, l3b, out_h3);
  (void)in_sizes; (void)n_in; (void)out_size; (void)ws_size;
}
